// Top1Router_49520972923488
// MI455X (gfx1250) — matrix operand order repaired and checked by reading, not compiled, not run on hardware
//
#include <hip/hip_runtime.h>

// ---------------------------------------------------------------------------
// Top-1 MoE router for MI455X (gfx1250, wave32, WMMA bf16 16x16x32).
//   logits = x @ W^T + b ; softmax ; top1/w ; lb_loss ; z_loss
// x: [16384, 2048] f32, W: [64, 2048] f32, b: [64] f32
// out: [ top1(16384) | w(16384) | lb_loss(1) | z_loss(1) ]  (floats)
// ---------------------------------------------------------------------------

typedef __attribute__((ext_vector_type(16))) __bf16 v16bf;
typedef __attribute__((ext_vector_type(8)))  float  v8f;

#define D_MODEL    2048
#define N_EXP      64
#define N_TOK      16384
#define KCHUNK     64          // K elements staged per iteration (2 WMMA k-steps)
#define TOK_PER_WG 128         // 8 waves * 16 tokens
#define THREADS    256
#define LDS_STRIDE 72          // bf16 elements per LDS row (64 + 8 pad, 16B aligned)

// fp32 -> bf16, round-to-nearest-even
__device__ __forceinline__ unsigned short f2bf(float f) {
  unsigned u = __builtin_bit_cast(unsigned, f);
  unsigned r = (u + 0x7FFFu + ((u >> 16) & 1u)) >> 16;
  return (unsigned short)r;
}

union Frag32B {             // 32-byte WMMA operand (16 x bf16)
  uint4 u[2];
  v16bf v;
};

// ---------------------------------------------------------------------------
// Kernel 0: convert W to bf16 (L2-resident, reused by every WG) + zero the
// loss accumulators (probsum[64] | loadsum[64] | zsum[1]).
// ---------------------------------------------------------------------------
__global__ void prep_kernel(const float* __restrict__ W,
                            unsigned short* __restrict__ wbf,
                            float* __restrict__ acc) {
  int i = blockIdx.x * blockDim.x + threadIdx.x;
  if (i < N_EXP * D_MODEL) wbf[i] = f2bf(W[i]);
  if (blockIdx.x == 0 && threadIdx.x < 129) acc[threadIdx.x] = 0.0f;
}

// ---------------------------------------------------------------------------
// Kernel 1: main router. One wave = 16 tokens x 64 experts.
// ---------------------------------------------------------------------------
__global__ __launch_bounds__(THREADS) void router_kernel(
    const float* __restrict__ x,
    const unsigned short* __restrict__ wbf,
    const float* __restrict__ b,
    float* __restrict__ out,
    float* __restrict__ acc) {

  __shared__ __align__(16) unsigned short xs[TOK_PER_WG * LDS_STRIDE]; // 18 KB

  const int tid     = threadIdx.x;
  const int lane    = tid & 31;
  const int wave    = tid >> 5;        // 0..7
  const int l16     = lane & 15;       // column within 16-wide tile
  const int hi      = lane >> 4;       // half-wave selector (A/B K halves)
  const int tokBase = blockIdx.x * TOK_PER_WG;
  const int waveTok = tokBase + wave * 16;

  const v8f zero = {0.f, 0.f, 0.f, 0.f, 0.f, 0.f, 0.f, 0.f};
  v8f c[4];
#pragma unroll
  for (int t = 0; t < 4; ++t) c[t] = zero;

  for (int k0 = 0; k0 < D_MODEL; k0 += KCHUNK) {
    // ---- cooperative stage: x[tokBase..+128][k0..k0+64) f32 -> bf16 LDS ----
#pragma unroll
    for (int j = 0; j < 8; ++j) {
      int s   = j * THREADS + tid;     // 2048 float4 slots
      int tok = s >> 4;                // 16 float4 per token row
      int kq  = s & 15;
      const float4 f =
          *(const float4*)(x + (size_t)(tokBase + tok) * D_MODEL + k0 + kq * 4);
      uint2 pk;
      pk.x = (unsigned)f2bf(f.x) | ((unsigned)f2bf(f.y) << 16);
      pk.y = (unsigned)f2bf(f.z) | ((unsigned)f2bf(f.w) << 16);
      *(uint2*)(xs + tok * LDS_STRIDE + kq * 4) = pk;
    }
    // prefetch next x chunk while we compute on this one
    if (k0 + KCHUNK < D_MODEL) {
      int tok = tid >> 4, kq = tid & 15;
      __builtin_prefetch(
          x + (size_t)(tokBase + tok) * D_MODEL + (k0 + KCHUNK) + kq * 4, 0, 0);
    }
    __syncthreads();

    // ---- 2 k-steps of 32, 4 N-tiles each: 8 WMMAs per wave per chunk ----
#pragma unroll
    for (int ks = 0; ks < 2; ++ks) {
      // A fragment (16x32 bf16): lane<16 holds K[0..15]; lane>=16
      // holds K[16..31] of its token row.
      Frag32B a;
      const unsigned short* arow =
          xs + (wave * 16 + l16) * LDS_STRIDE + ks * 32 + hi * 16;
      a.u[0] = *(const uint4*)(arow);
      a.u[1] = *(const uint4*)(arow + 8);    // +8 halfs = +16 bytes
#pragma unroll
      for (int t = 0; t < 4; ++t) {
        // B fragment (32x16 bf16): lane's column = expert row of W;
        // 16 contiguous K values straight from L2-resident bf16 W.
        Frag32B bf;
        const uint4* wp = (const uint4*)(wbf +
            (size_t)(t * 16 + l16) * D_MODEL + k0 + ks * 32 + hi * 16);
        bf.u[0] = wp[0];
        bf.u[1] = wp[1];
        c[t] = __builtin_amdgcn_wmma_f32_16x16x32_bf16(
            false, a.v, false, bf.v, (short)0, c[t], false, false);
      }
    }
    __syncthreads();
  }

  // ------------------- fused epilogue -------------------
  float* probsum = acc;            // [64]
  float* loadsum = acc + 64;       // [64]
  float* zsum    = acc + 128;      // [1]

  float bias[4];
#pragma unroll
  for (int t = 0; t < 4; ++t) bias[t] = b[t * 16 + l16];

  float pacc[4] = {0.f, 0.f, 0.f, 0.f};

#pragma unroll
  for (int r = 0; r < 8; ++r) {
    // C layout: VGPR r holds row M=r (lanes 0-15) / M=r+8 (lanes 16-31),
    // column = 16*t + l16.
    float v[4];
#pragma unroll
    for (int t = 0; t < 4; ++t) v[t] = c[t][r] + bias[t];

    // argmax (lowest index on ties, matching jnp.argmax)
    float mval = v[0];
    int   midx = l16;
#pragma unroll
    for (int t = 1; t < 4; ++t) {
      if (v[t] > mval) { mval = v[t]; midx = t * 16 + l16; }
    }
#pragma unroll
    for (int m = 8; m >= 1; m >>= 1) {
      float ov = __shfl_xor(mval, m);
      int   oi = __shfl_xor(midx, m);
      if (ov > mval || (ov == mval && oi < midx)) { mval = ov; midx = oi; }
    }

    // sum of exp (shifted by row max)
    float p[4], s = 0.f;
#pragma unroll
    for (int t = 0; t < 4; ++t) { p[t] = __expf(v[t] - mval); s += p[t]; }
#pragma unroll
    for (int m = 8; m >= 1; m >>= 1) s += __shfl_xor(s, m);

    float inv = 1.0f / s;
#pragma unroll
    for (int t = 0; t < 4; ++t) pacc[t] += p[t] * inv;   // per-expert prob acc

    if (l16 == r) {                       // one writer per token row
      int token = waveTok + r + hi * 8;
      out[token]         = (float)midx;   // top1 (as float)
      out[N_TOK + token] = inv;           // w = exp(0)/sumexp
      float lse = mval + __logf(s);
      atomicAdd(zsum, lse * lse);
      atomicAdd(&loadsum[midx], 1.0f);
    }
  }

  // fold the two half-waves, one atomic per expert column per wave
#pragma unroll
  for (int t = 0; t < 4; ++t) {
    float agg = pacc[t] + __shfl_xor(pacc[t], 16);
    if (hi == 0) atomicAdd(&probsum[t * 16 + l16], agg);
  }
}

// ---------------------------------------------------------------------------
// Kernel 2: scalars.  lb = E * sum(importance * load),  z = mean(lse^2)
// ---------------------------------------------------------------------------
__global__ void finalize_kernel(const float* __restrict__ acc,
                                float* __restrict__ out) {
  int t = threadIdx.x;  // 32 threads
  const float* probsum = acc;
  const float* loadsum = acc + 64;
  float prod = probsum[t] * loadsum[t] + probsum[t + 32] * loadsum[t + 32];
#pragma unroll
  for (int m = 16; m >= 1; m >>= 1) prod += __shfl_xor(prod, m);
  if (t == 0) {
    const float invN = 1.0f / (float)N_TOK;
    out[2 * N_TOK]     = (float)N_EXP * prod * invN * invN;  // lb_loss
    out[2 * N_TOK + 1] = acc[128] * invN;                    // z_loss
  }
}

// ---------------------------------------------------------------------------
extern "C" void kernel_launch(void* const* d_in, const int* in_sizes, int n_in,
                              void* d_out, int out_size, void* d_ws,
                              size_t ws_size, hipStream_t stream) {
  const float* x = (const float*)d_in[0];   // [16384, 2048]
  const float* W = (const float*)d_in[1];   // [64, 2048]
  const float* b = (const float*)d_in[2];   // [64]
  float* out = (float*)d_out;

  unsigned short* wbf = (unsigned short*)d_ws;                      // 256 KB
  float* acc = (float*)((char*)d_ws + (size_t)N_EXP * D_MODEL * 2); // 129 f32

  prep_kernel<<<(N_EXP * D_MODEL + THREADS - 1) / THREADS, THREADS, 0, stream>>>(
      W, wbf, acc);
  router_kernel<<<N_TOK / TOK_PER_WG, THREADS, 0, stream>>>(x, wbf, b, out, acc);
  finalize_kernel<<<1, 32, 0, stream>>>(acc, out);
}
